// GAT_test_81509889344204
// MI455X (gfx1250) — matrix operand order repaired and checked by reading, not compiled, not run on hardware
//
#include <hip/hip_runtime.h>
#include <hip/hip_bf16.h>
#include <math.h>

typedef __attribute__((ext_vector_type(16))) __bf16 v16bf;
typedef __attribute__((ext_vector_type(8)))  __bf16 v8bf;
typedef __attribute__((ext_vector_type(8)))  float  v8f;

#define HEADS 8
#define NGRAPH 128
#define EDIM 147
#define KBLK 128                      // K-slice staged in LDS
#define LDB  (KBLK + 8)               // padded LDS stride: conflict-free fragment reads

// ---------- helpers ----------
__device__ __forceinline__ unsigned f2s(float f) {           // sortable-uint encode
  unsigned u = __float_as_uint(f);
  return (u & 0x80000000u) ? ~u : (u | 0x80000000u);
}
__device__ __forceinline__ float s2f(unsigned s) {           // decode
  unsigned u = (s & 0x80000000u) ? (s & 0x7FFFFFFFu) : ~s;
  return __uint_as_float(u);
}

// ---------- conversion ----------
__global__ void k_f32_to_bf16(const float* __restrict__ in, __bf16* __restrict__ out, int n) {
  int i = blockIdx.x * blockDim.x + threadIdx.x;
  if (i < n) out[i] = (__bf16)in[i];
}

// ---------- WMMA GEMM: C[M,HC] = A[M,K](bf16) x W[HC,K](bf16)^T, f32 accum ----------
// Block = 8 waves; each wave owns a 16-row tile, all waves share a 64-col group.
// B tile (64 W-rows) is staged per-block in LDS in K-slices of 128.
__global__ __launch_bounds__(256)
void k_gemm_wmma(const __bf16* __restrict__ A, const __bf16* __restrict__ W,
                 float* __restrict__ Cmat, int Mrows, int K, int HC) {
  __shared__ __bf16 Bs[64 * LDB];                  // ~17 KB

  const int lane = threadIdx.x & 31;
  const int wave = threadIdx.x >> 5;
  const int l    = lane & 15;
  const int hi   = lane >> 4;
  const int mtile = blockIdx.x * 8 + wave;
  const int row0  = mtile * 16;
  const int valid = (row0 < Mrows);                // wave-uniform
  const int ncb   = blockIdx.y * 64;

  union { v16bf v; v8bf h[2]; } a, b;
  v8f acc[4] = {};

  const __bf16* Arow = A + (size_t)(row0 + l) * K;
  // per-wave fragment base pointers into LDS (column n = nt*16 + l)
  const __bf16* Bb[4];
#pragma unroll
  for (int nt = 0; nt < 4; ++nt) Bb[nt] = &Bs[(nt * 16 + l) * LDB + hi * 16];

  for (int kb = 0; kb < K; kb += KBLK) {
    const int kw  = (K - kb < KBLK) ? (K - kb) : KBLK;
    const int nch = kw >> 3;                       // 16B chunks per column
    __syncthreads();                               // previous slice fully consumed
    for (int idx = threadIdx.x; idx < 64 * nch; idx += 256) {
      int col = idx / nch;
      int kc  = (idx - col * nch) << 3;
      *(v8bf*)&Bs[col * LDB + kc] =
          *(const v8bf*)(W + (size_t)(ncb + col) * K + kb + kc);
    }
    __syncthreads();

    if (valid) {
      for (int k0 = 0; k0 < kw; k0 += 32) {
        __builtin_prefetch(Arow + kb + k0 + 256, 0, 1);   // global_prefetch_b8
        // A fragment (16x32 bf16): row M=l, K = hi*16..+15
        const v8bf* Ap = (const v8bf*)(Arow + kb + k0);
        a.h[0] = Ap[2 * hi];
        a.h[1] = Ap[2 * hi + 1];
#pragma unroll
        for (int nt = 0; nt < 4; ++nt) {
          // B fragment (32x16 bf16) from LDS: col n=l(+nt*16), K = k0 + hi*16 + i
          b.h[0] = *(const v8bf*)(Bb[nt] + k0);
          b.h[1] = *(const v8bf*)(Bb[nt] + k0 + 8);
          acc[nt] = __builtin_amdgcn_wmma_f32_16x16x32_bf16(
              false, a.v, false, b.v, (short)0, acc[nt], false, false);
        }
      }
    }
  }

  if (valid) {
#pragma unroll
    for (int nt = 0; nt < 4; ++nt)
#pragma unroll
      for (int r = 0; r < 8; ++r)
        Cmat[(size_t)(row0 + hi * 8 + r) * HC + ncb + nt * 16 + l] = acc[nt][r];
  }
}

// ---------- self-loop edge attr (mean of incoming) ----------
__global__ void k_deg(const int* __restrict__ dst, float* __restrict__ deg, int E) {
  int i = blockIdx.x * blockDim.x + threadIdx.x;
  if (i < E) atomicAdd(&deg[dst[i]], 1.0f);
}
__global__ void k_loop_sum(const int* __restrict__ dst, const float* __restrict__ ea,
                           float* __restrict__ la, int E) {
  int i = blockIdx.x * blockDim.x + threadIdx.x;
  if (i >= E * EDIM) return;
  int e = i / EDIM, d = i - e * EDIM;
  atomicAdd(&la[(size_t)dst[e] * EDIM + d], ea[(size_t)e * EDIM + d]);
}
__global__ void k_loop_div(float* __restrict__ la, const float* __restrict__ deg, int N) {
  int i = blockIdx.x * blockDim.x + threadIdx.x;
  if (i >= N * EDIM) return;
  la[i] /= fmaxf(deg[i / EDIM], 1.0f);
}

// ---------- attention pieces ----------
__global__ void k_alpha_nodes(const float* __restrict__ xp, const float* __restrict__ as_,
                              const float* __restrict__ ad_, float* __restrict__ als,
                              float* __restrict__ ald, int N, int C, int HC) {
  int i = blockIdx.x * blockDim.x + threadIdx.x;
  if (i >= N * HEADS) return;
  int n = i / HEADS, h = i - n * HEADS;
  const float* row = xp + (size_t)n * HC + h * C;
  float ss = 0.f, sd = 0.f;
  for (int c = 0; c < C; ++c) { float v = row[c]; ss += v * as_[h * C + c]; sd += v * ad_[h * C + c]; }
  als[i] = ss; ald[i] = sd;
}
// V[h,d] = sum_c We[h*C+c, d] * ae[h*C+c]   (folds the edge projection)
__global__ void k_V(const float* __restrict__ We, const float* __restrict__ ae,
                    float* __restrict__ V, int C) {
  int i = blockIdx.x * blockDim.x + threadIdx.x;
  if (i >= HEADS * EDIM) return;
  int h = i / EDIM, d = i - h * EDIM;
  float s = 0.f;
  for (int c = 0; c < C; ++c) s += We[(size_t)(h * C + c) * EDIM + d] * ae[h * C + c];
  V[i] = s;
}
__global__ void k_alpha_e(const float* __restrict__ ea, const float* __restrict__ la,
                          const float* __restrict__ V, float* __restrict__ aeb, int E, int Ef) {
  int i = blockIdx.x * blockDim.x + threadIdx.x;
  if (i >= Ef * HEADS) return;
  int e = i / HEADS, h = i - e * HEADS;
  const float* row = (e < E) ? (ea + (size_t)e * EDIM) : (la + (size_t)(e - E) * EDIM);
  const float* v = V + h * EDIM;
  float s = 0.f;
  for (int d = 0; d < EDIM; ++d) s += row[d] * v[d];
  aeb[i] = s;
}
__global__ void k_logits_max(const int* __restrict__ src, const int* __restrict__ dst,
                             const float* __restrict__ als, const float* __restrict__ ald,
                             const float* __restrict__ aeb, float* __restrict__ lg,
                             unsigned* __restrict__ mx, int E, int Ef) {
  int i = blockIdx.x * blockDim.x + threadIdx.x;
  if (i >= Ef * HEADS) return;
  int e = i / HEADS, h = i - e * HEADS;
  int s = (e < E) ? src[e] : (e - E);
  int d = (e < E) ? dst[e] : (e - E);
  float v = als[s * HEADS + h] + ald[d * HEADS + h] + aeb[i];
  v = (v > 0.f) ? v : 0.2f * v;                    // leaky_relu(0.2)
  lg[i] = v;
  atomicMax(&mx[d * HEADS + h], f2s(v));
}
__global__ void k_mx_fix(unsigned* __restrict__ mxu, float* __restrict__ mxf, int n) {
  int i = blockIdx.x * blockDim.x + threadIdx.x;
  if (i >= n) return;
  float m = s2f(mxu[i]);
  mxf[i] = isfinite(m) ? m : 0.0f;
}
__global__ void k_exp_denom(const int* __restrict__ dst, float* __restrict__ lg,
                            const float* __restrict__ mxf, float* __restrict__ dnm,
                            int E, int Ef) {
  int i = blockIdx.x * blockDim.x + threadIdx.x;
  if (i >= Ef * HEADS) return;
  int e = i / HEADS, h = i - e * HEADS;
  int d = (e < E) ? dst[e] : (e - E);
  float ex = __expf(lg[i] - mxf[d * HEADS + h]);
  lg[i] = ex;
  atomicAdd(&dnm[d * HEADS + h], ex);
}
__global__ void k_scatter(const int* __restrict__ src, const int* __restrict__ dst,
                          const float* __restrict__ lg, const float* __restrict__ dnm,
                          const float* __restrict__ xp, float* __restrict__ outb,
                          int E, int Ef, int C, int HC) {
  int i = blockIdx.x * blockDim.x + threadIdx.x;
  if (i >= Ef * HEADS) return;
  int e = i / HEADS, h = i - e * HEADS;
  int s = (e < E) ? src[e] : (e - E);
  int d = (e < E) ? dst[e] : (e - E);
  float attn = lg[i] / (dnm[d * HEADS + h] + 1e-16f);
  const float* xr = xp + (size_t)s * HC + h * C;
  float* orow = outb + (size_t)d * HC + h * C;
  for (int c = 0; c < C; ++c) atomicAdd(&orow[c], xr[c] * attn);
}
__global__ void k_bias_act(float* __restrict__ outb, const float* __restrict__ b,
                           int total, int HC, int do_elu) {
  int i = blockIdx.x * blockDim.x + threadIdx.x;
  if (i >= total) return;
  float v = outb[i] + b[i % HC];
  if (do_elu) v = (v > 0.f) ? v : (__expf(v) - 1.0f);
  outb[i] = v;
}

// ---------- pooling + classifier head ----------
__global__ void k_pool(const float* __restrict__ h, const int* __restrict__ batch,
                       float* __restrict__ pooled, float* __restrict__ cnt, int N) {
  int n = blockIdx.x * blockDim.x + threadIdx.x;
  if (n >= N) return;
  int g = batch[n];
  atomicAdd(&cnt[g], 1.0f);
  const float* row = h + (size_t)n * 256;
  float* prow = pooled + (size_t)g * 256;
  for (int j = 0; j < 256; ++j) atomicAdd(&prow[j], row[j]);
}
__global__ void k_head(const float* __restrict__ pooled, const float* __restrict__ cnt,
                       const float* __restrict__ Wc, const float* __restrict__ bc,
                       float* __restrict__ out, int NG) {
  int i = blockIdx.x * blockDim.x + threadIdx.x;
  if (i >= NG * 32) return;
  int g = i / 32, o = i - g * 32;
  float inv = 1.0f / fmaxf(cnt[g], 1.0f);
  const float* prow = pooled + (size_t)g * 256;
  const float* wrow = Wc + (size_t)o * 256;
  float s = 0.f;
  for (int j = 0; j < 256; ++j) s += prow[j] * inv * wrow[j];
  out[i] = s + bc[o];
}

extern "C" void kernel_launch(void* const* d_in, const int* in_sizes, int n_in,
                              void* d_out, int out_size, void* d_ws, size_t ws_size,
                              hipStream_t stream) {
  (void)n_in; (void)out_size; (void)ws_size;
  const int N  = in_sizes[0] / 64;        // 50000
  const int E  = in_sizes[22] / 2;        // 200000
  const int Ef = E + N;                   // 250000

  const float* x     = (const float*)d_in[0];
  const float* ea    = (const float*)d_in[1];
  const float* Wc    = (const float*)d_in[20];
  const float* bc    = (const float*)d_in[21];
  const int*   ei    = (const int*)d_in[22];
  const int*   srcv  = ei;
  const int*   dstv  = ei + E;
  const int*   batch = (const int*)d_in[23];

  // workspace carve-up (256B aligned)
  char* base = (char*)d_ws; size_t off = 0;
  auto carve = [&](size_t bytes) -> char* {
    char* p = base + off; off += (bytes + 255) & ~(size_t)255; return p;
  };
  __bf16* hbf  = (__bf16*)carve((size_t)N * 512 * 2);
  __bf16* wbf  = (__bf16*)carve((size_t)512 * 512 * 2);
  float* xp    = (float*)carve((size_t)N * 512 * 4);
  float* buf0  = (float*)carve((size_t)N * 512 * 4);
  float* buf1  = (float*)carve((size_t)N * 512 * 4);
  float* als   = (float*)carve((size_t)N * HEADS * 4);
  float* ald   = (float*)carve((size_t)N * HEADS * 4);
  float* Vb    = (float*)carve((size_t)HEADS * EDIM * 4);
  float* aeb   = (float*)carve((size_t)Ef * HEADS * 4);
  float* lg    = (float*)carve((size_t)Ef * HEADS * 4);
  float* mx    = (float*)carve((size_t)N * HEADS * 4);
  float* dnm   = (float*)carve((size_t)N * HEADS * 4);
  float* deg   = (float*)carve((size_t)N * 4);
  float* la    = (float*)carve((size_t)N * EDIM * 4);
  float* pooled= (float*)carve((size_t)NGRAPH * 256 * 4);
  float* cnt   = (float*)carve((size_t)NGRAPH * 4);

  const int B = 256;
  auto nb = [&](long long t) { return (unsigned)((t + B - 1) / B); };

  // self-loop mean edge attrs
  hipMemsetAsync(deg, 0, (size_t)N * 4, stream);
  hipMemsetAsync(la, 0, (size_t)N * EDIM * 4, stream);
  k_deg<<<nb(E), B, 0, stream>>>(dstv, deg, E);
  k_loop_sum<<<nb((long long)E * EDIM), B, 0, stream>>>(dstv, ea, la, E);
  k_loop_div<<<nb((long long)N * EDIM), B, 0, stream>>>(la, deg, N);

  const int cins[3] = {64, 512, 512};
  const int Cs[3]   = {64, 64, 32};
  float* outs[3] = {buf0, buf1, buf0};
  const float* hin = x;

  for (int L = 0; L < 3; ++L) {
    const int cin = cins[L], C = Cs[L], HC = HEADS * C;
    const float* Wf  = (const float*)d_in[2 + 6 * L + 0];
    const float* Wef = (const float*)d_in[2 + 6 * L + 1];
    const float* asf = (const float*)d_in[2 + 6 * L + 2];
    const float* adf = (const float*)d_in[2 + 6 * L + 3];
    const float* aef = (const float*)d_in[2 + 6 * L + 4];
    const float* bf  = (const float*)d_in[2 + 6 * L + 5];
    float* outb = outs[L];

    k_f32_to_bf16<<<nb((long long)N * cin), B, 0, stream>>>(hin, hbf, N * cin);
    k_f32_to_bf16<<<nb((long long)HC * cin), B, 0, stream>>>(Wf, wbf, HC * cin);

    {
      int mtiles = (N + 15) / 16;
      dim3 g((unsigned)((mtiles + 7) / 8), (unsigned)(HC / 64));
      k_gemm_wmma<<<g, 256, 0, stream>>>(hbf, wbf, xp, N, cin, HC);
    }

    k_alpha_nodes<<<nb((long long)N * HEADS), B, 0, stream>>>(xp, asf, adf, als, ald, N, C, HC);
    k_V<<<nb((long long)HEADS * EDIM), B, 0, stream>>>(Wef, aef, Vb, C);
    k_alpha_e<<<nb((long long)Ef * HEADS), B, 0, stream>>>(ea, la, Vb, aeb, E, Ef);

    hipMemsetAsync(mx, 0, (size_t)N * HEADS * 4, stream);   // sortable 0 == floor
    hipMemsetAsync(dnm, 0, (size_t)N * HEADS * 4, stream);
    hipMemsetAsync(outb, 0, (size_t)N * HC * 4, stream);

    k_logits_max<<<nb((long long)Ef * HEADS), B, 0, stream>>>(srcv, dstv, als, ald, aeb, lg,
                                                              (unsigned*)mx, E, Ef);
    k_mx_fix<<<nb((long long)N * HEADS), B, 0, stream>>>((unsigned*)mx, mx, N * HEADS);
    k_exp_denom<<<nb((long long)Ef * HEADS), B, 0, stream>>>(dstv, lg, mx, dnm, E, Ef);
    k_scatter<<<nb((long long)Ef * HEADS), B, 0, stream>>>(srcv, dstv, lg, dnm, xp, outb, E, Ef, C, HC);
    k_bias_act<<<nb((long long)N * HC), B, 0, stream>>>(outb, bf, N * HC, HC, (L < 2) ? 1 : 0);
    hin = outb;
  }

  hipMemsetAsync(pooled, 0, (size_t)NGRAPH * 256 * 4, stream);
  hipMemsetAsync(cnt, 0, (size_t)NGRAPH * 4, stream);
  k_pool<<<nb(N), B, 0, stream>>>(buf0, batch, pooled, cnt, N);
  k_head<<<nb((long long)NGRAPH * 32), B, 0, stream>>>(pooled, cnt, Wc, bc, (float*)d_out, NGRAPH);
}
